// VectorQuantizer_62878321214379
// MI455X (gfx1250) — matrix operand order repaired and checked by reading, not compiled, not run on hardware
//
#include <hip/hip_runtime.h>
#include <hip/hip_bf16.h>

typedef __attribute__((ext_vector_type(16))) _Float16 v16h;
typedef __attribute__((ext_vector_type(8)))  float    v8f;

#define D_DIM   256
#define K_CODES 1024
#define N_ROWS  32768            // 32 * 32 * 32 spatial positions
#define TM      128              // rows per block in argmin kernel (8 waves x 16)
#define LDSP    264              // padded row stride in halves (528B: 16B aligned, bank-spread)
#define OUT_ELEMS 8388608ULL     // 32*256*32*32

// ---------------------------------------------------------------------------
// Kernel 1: codebook f32 -> f16 copy + exact f32 row norms
// grid = 1024 blocks (one per code), block = 256 threads (one per channel)
// ---------------------------------------------------------------------------
__global__ __launch_bounds__(256) void vq_prep(const float* __restrict__ cb,
                                               _Float16* __restrict__ cbh,
                                               float* __restrict__ cbn) {
  const int k = blockIdx.x;
  const int t = threadIdx.x;
  const float v = cb[(size_t)k * D_DIM + t];
  cbh[(size_t)k * D_DIM + t] = (_Float16)v;

  float s = v * v;
  #pragma unroll
  for (int off = 16; off; off >>= 1) s += __shfl_xor(s, off, 32);

  __shared__ float ws[8];
  if ((t & 31) == 0) ws[t >> 5] = s;
  __syncthreads();
  if (t == 0) {
    float tot = 0.f;
    #pragma unroll
    for (int i = 0; i < 8; ++i) tot += ws[i];
    cbn[k] = tot;
  }
}

// ---------------------------------------------------------------------------
// Kernel 2: fused GEMM + argmin via v_wmma_f32_16x16x32_f16
// grid = N_ROWS/TM = 256 blocks, block = 256 threads (8 wave32; 16 rows/wave)
// metric m_k = ||e_k||^2 - 2 * dot(z_n, e_k)   (||z||^2 constant per row)
// Codebook tiles are double-buffered in LDS: one barrier per 16-code tile,
// global fetch of tile kt+1 overlaps the 8 WMMAs of tile kt.
// ---------------------------------------------------------------------------
__global__ __launch_bounds__(256) void vq_argmin(const float* __restrict__ z,
                                                 const _Float16* __restrict__ cbh,
                                                 const float* __restrict__ cbn,
                                                 int* __restrict__ idx_out) {
  __shared__ __align__(16) _Float16 lds_z[TM][LDSP];      // 67,584 B
  __shared__ __align__(16) _Float16 lds_cb[2][16][LDSP];  // 16,896 B
  __shared__ float lds_n[K_CODES];                        //  4,096 B

  const int t  = threadIdx.x;
  const int n0 = blockIdx.x * TM;
  const int b  = n0 >> 10;        // batch (H*W = 1024)
  const int hw0 = n0 & 1023;

  // ---- stage z tile: z[b, c, hw0 + nn] -> lds_z[nn][c] (f32 -> f16) ----
  #pragma unroll 4
  for (int i = 0; i < (TM * D_DIM) / 256; ++i) {   // 128 iterations
    const int flat = i * 256 + t;
    const int c  = flat >> 7;       // 0..255
    const int nn = flat & 127;      // 0..127
    const float v = z[((size_t)(b * D_DIM + c) << 10) + hw0 + nn];
    lds_z[nn][c] = (_Float16)v;
  }
  // ---- stage codebook norms ----
  #pragma unroll
  for (int i = 0; i < K_CODES / 256; ++i) lds_n[i * 256 + t] = cbn[i * 256 + t];

  // ---- stage first codebook tile into buffer 0 (512 x 16B / 256 threads) ----
  const int r16 = t >> 5;          // 0..7
  const int c16 = t & 31;          // 16B chunk within 512B row
  *(uint4*)&lds_cb[0][r16][c16 * 8] =
      *(const uint4*)(cbh + ((size_t)r16 << 8) + c16 * 8);
  *(uint4*)&lds_cb[0][r16 + 8][c16 * 8] =
      *(const uint4*)(cbh + ((size_t)(r16 + 8) << 8) + c16 * 8);
  __syncthreads();

  const int lane = t & 31;
  const int wv   = t >> 5;          // wave id 0..7, owns rows [wv*16, wv*16+16)
  const int row  = lane & 15;       // A: M index / B: N index / C: column
  const int half = lane >> 4;       // lane half selector

  // ---- hoist A fragments for all 8 K-chunks (16-bit A layout, wave32) ----
  // lanes 0-15: K = d0+{0..7}, d0+{16..23};  lanes 16-31: K = d0+{8..15}, d0+{24..31}
  v16h afrag[8];
  {
    const _Float16* zr = &lds_z[wv * 16 + row][0];
    #pragma unroll
    for (int ch = 0; ch < 8; ++ch) {
      const uint4* p = (const uint4*)(zr + ch * 32 + half * 8);
      *((uint4*)&afrag[ch])     = p[0];   // 8 halves
      *((uint4*)&afrag[ch] + 1) = p[2];   // +16 halves (32B ahead)
    }
  }

  float bestm[8];
  int   bestk[8];
  #pragma unroll
  for (int r = 0; r < 8; ++r) { bestm[r] = 3.4e38f; bestk[r] = 0; }

  for (int kt = 0; kt < K_CODES / 16; ++kt) {      // 64 code tiles
    const int cur = kt & 1;
    const bool more = (kt + 1 < K_CODES / 16);

    // ---- fetch NEXT tile into registers (overlaps the WMMAs below) ----
    uint4 nx0, nx1;
    if (more) {
      const _Float16* src = cbh + ((size_t)((kt + 1) * 16) << 8);
      nx0 = *(const uint4*)(src + ((size_t)r16 << 8) + c16 * 8);
      nx1 = *(const uint4*)(src + ((size_t)(r16 + 8) << 8) + c16 * 8);
    }
    if (kt + 2 < K_CODES / 16)      // hint tile kt+2 toward L2 (global_prefetch_b8)
      __builtin_prefetch(cbh + ((size_t)((kt + 2) * 16) << 8) + lane * 16, 0, 1);

    // ---- 16x16 output tile, accumulate over D = 8 chunks of 32 ----
    v8f acc = {};
    #pragma unroll
    for (int ch = 0; ch < 8; ++ch) {
      // B layout: lanes 0-15 hold K = d0+{0..7},{16..23} (col N = lane), lanes 16-31 K = d0+{8..15},{24..31}
      v16h bfrag;
      const uint4* p = (const uint4*)(&lds_cb[cur][row][ch * 32 + half * 8]);
      *((uint4*)&bfrag)     = p[0];
      *((uint4*)&bfrag + 1) = p[2];
      acc = __builtin_amdgcn_wmma_f32_16x16x32_f16(
          false, afrag[ch], false, bfrag, (short)0, acc, false, false);
    }

    // ---- fused argmin update: lane's column = code (kt*16 + row) ----
    const int   k  = kt * 16 + row;
    const float nk = lds_n[k];
    #pragma unroll
    for (int r = 0; r < 8; ++r) {
      const float m = fmaf(-2.0f, acc[r], nk);
      const bool better = (m < bestm[r]);
      bestk[r] = better ? k : bestk[r];
      bestm[r] = better ? m : bestm[r];
    }

    // ---- commit NEXT tile into the other buffer, then one barrier ----
    if (more) {
      *(uint4*)&lds_cb[cur ^ 1][r16][c16 * 8]     = nx0;
      *(uint4*)&lds_cb[cur ^ 1][r16 + 8][c16 * 8] = nx1;
    }
    __syncthreads();
  }

  // ---- reduce (min, first-index) across the 16 lanes of each half ----
  #pragma unroll
  for (int r = 0; r < 8; ++r) {
    #pragma unroll
    for (int off = 8; off; off >>= 1) {
      const float om = __shfl_xor(bestm[r], off, 32);
      const int   ok = __shfl_xor(bestk[r], off, 32);
      if (om < bestm[r] || (om == bestm[r] && ok < bestk[r])) {
        bestm[r] = om; bestk[r] = ok;
      }
    }
    if (row == 0) {   // lanes 0 and 16: rows r and 8+r of this wave's M tile
      idx_out[n0 + wv * 16 + half * 8 + r] = bestk[r];
    }
  }
}

// ---------------------------------------------------------------------------
// Kernel 3: gather zq = codebook[idx] in [B,C,H,W] order + per-block partial
//           of sum((z - zq)^2).  grid = 32768, block = 256.
// ---------------------------------------------------------------------------
__global__ __launch_bounds__(256) void vq_gather(const float* __restrict__ z,
                                                 const float* __restrict__ cb,
                                                 const int* __restrict__ idx,
                                                 float* __restrict__ outq,
                                                 float* __restrict__ partial) {
  const size_t f = (size_t)blockIdx.x * 256 + threadIdx.x;  // < 8388608
  const int hw = (int)(f & 1023);
  const int c  = (int)((f >> 10) & 255);
  const int b  = (int)(f >> 18);
  const int n  = (b << 10) + hw;

  const int   k  = idx[n];
  const float q  = cb[((size_t)k << 8) + c];
  const float zv = z[f];
  outq[f] = q;

  float d = zv - q;
  d *= d;
  #pragma unroll
  for (int off = 16; off; off >>= 1) d += __shfl_xor(d, off, 32);

  __shared__ float ws[8];
  if ((threadIdx.x & 31) == 0) ws[threadIdx.x >> 5] = d;
  __syncthreads();
  if (threadIdx.x == 0) {
    float s = 0.f;
    #pragma unroll
    for (int i = 0; i < 8; ++i) s += ws[i];
    partial[blockIdx.x] = s;
  }
}

// ---------------------------------------------------------------------------
// Kernel 4: deterministic fixed-order reduction of partials -> loss scalar
// ---------------------------------------------------------------------------
__global__ __launch_bounds__(256) void vq_finalize(const float* __restrict__ partial,
                                                   int np, float* __restrict__ loss) {
  __shared__ float ws[256];
  float s = 0.f;
  for (int i = threadIdx.x; i < np; i += 256) s += partial[i];
  ws[threadIdx.x] = s;
  __syncthreads();
  for (int off = 128; off; off >>= 1) {
    if (threadIdx.x < off) ws[threadIdx.x] += ws[threadIdx.x + off];
    __syncthreads();
  }
  if (threadIdx.x == 0) loss[0] = 1.25f * ws[0] / (float)OUT_ELEMS;
}

// ---------------------------------------------------------------------------
extern "C" void kernel_launch(void* const* d_in, const int* in_sizes, int n_in,
                              void* d_out, int out_size, void* d_ws, size_t ws_size,
                              hipStream_t stream) {
  const float* z  = (const float*)d_in[0];   // [32,256,32,32]
  const float* cb = (const float*)d_in[1];   // [1024,256]

  float* out     = (float*)d_out;
  float* outq    = out;                      // 8,388,608 floats  (zq transposed)
  float* loss    = out + OUT_ELEMS;          // 1 float
  int*   idx_out = (int*)(out + OUT_ELEMS + 1);  // 32,768 ints

  char* ws = (char*)d_ws;
  _Float16* cbh   = (_Float16*)ws;                               // 512 KB
  float*    cbn   = (float*)(ws + (size_t)K_CODES * D_DIM * 2);  //   4 KB
  float*    parts = (float*)(ws + (size_t)K_CODES * D_DIM * 2 + K_CODES * 4); // 128 KB

  vq_prep<<<K_CODES, 256, 0, stream>>>(cb, cbh, cbn);
  vq_argmin<<<N_ROWS / TM, 256, 0, stream>>>(z, cbh, cbn, idx_out);
  vq_gather<<<(int)(OUT_ELEMS / 256), 256, 0, stream>>>(z, cb, idx_out, outq, parts);
  vq_finalize<<<1, 256, 0, stream>>>(parts, (int)(OUT_ELEMS / 256), loss);
}
